// GRULayer_18657337933909
// MI455X (gfx1250) — matrix operand order repaired and checked by reading, not compiled, not run on hardware
//
#include <hip/hip_runtime.h>
#include <stdint.h>

#define B_ 64
#define T_ 512
#define E_ 512
#define H_ 512
#define KTOT 1024   // E+H
#define NKT 32      // k-tiles of 32 over KTOT
#define NNT 32      // n-tiles of 16 over H
#define NWG_DIR 16

typedef __bf16 bf16;
typedef __attribute__((ext_vector_type(16))) __bf16   v16bf;
typedef __attribute__((ext_vector_type(8)))  float    v8f;
typedef __attribute__((ext_vector_type(4)))  uint32_t v4u;
typedef __attribute__((ext_vector_type(8)))  uint32_t v8u;

// ---- WMMA helpers -----------------------------------------------------------
// A-matrix 16x32 bf16 layout (ISA 7.12.2): lanes 0-15 hold row M=lane,
// K = k0+0..7 and k0+16..23; lanes 16-31 hold K = k0+8..15 and k0+24..31.
__device__ __forceinline__ v16bf load_a_bf16(const bf16* base, int stride, int lane, int k0) {
  int r  = lane & 15;
  int kk = k0 + ((lane >> 4) << 3);
  const bf16* p = base + (size_t)r * stride + kk;
  v4u a0 = *(const v4u*)(p);
  v4u a1 = *(const v4u*)(p + 16);
  v8u t  = __builtin_shufflevector(a0, a1, 0,1,2,3,4,5,6,7);
  return __builtin_bit_cast(v16bf, t);
}

// B-matrix 32x16 bf16, pre-packed so each lane's 16 values are contiguous
// (works for both global and LDS source pointers; LDS inferred -> ds_load_b128).
__device__ __forceinline__ v16bf load_b_pack(const bf16* pack, int lane) {
  const v4u* p = (const v4u*)(pack + lane * 16);
  v4u b0 = p[0], b1 = p[1];
  v8u t  = __builtin_shufflevector(b0, b1, 0,1,2,3,4,5,6,7);
  return __builtin_bit_cast(v16bf, t);
}

__device__ __forceinline__ v8f wmma_bf16(v16bf a, v16bf b, v8f c) {
  return __builtin_amdgcn_wmma_f32_16x16x32_bf16(false, a, false, b, (short)0, c, false, false);
}

__device__ __forceinline__ float sigmoidf_(float x) { return 1.0f / (1.0f + __expf(-x)); }

// ---- Weight pre-pack: W (H x KTOT fp32, row-major) -> bf16 B-tile layout ----
// pack[g][kt][ntg][lane][e] = W_g[ ntg*16 + (lane&15) ][ kt*32 + 8*(lane>>4) + (e&7) + 16*(e>>3) ]
__global__ void prep_pack(const float* __restrict__ Wz, const float* __restrict__ Wr,
                          const float* __restrict__ Wh, bf16* __restrict__ wpack) {
  size_t idx = (size_t)blockIdx.x * 256 + threadIdx.x;
  if (idx >= ((size_t)3 << 19)) return;
  int e    = idx & 15;
  int lane = (idx >> 4)  & 31;
  int ntg  = (idx >> 9)  & 31;
  int kt   = (idx >> 14) & 31;
  int g    = (int)(idx >> 19);
  int k = kt * 32 + ((lane >> 4) << 3) + (e & 7) + ((e >> 3) << 4);
  int n = ntg * 16 + (lane & 15);
  const float* W = (g == 0) ? Wz : (g == 1) ? Wr : Wh;
  wpack[idx] = (bf16)W[(size_t)n * KTOT + k];
}

// ---- Precompute x-part of all three gates: G[g][t][b][n] = x @ Wx.T + b -----
// Time-major G layout so the sequential scan reads contiguous 64KB per step.
__global__ void __launch_bounds__(256) precompute_x(
    const float* __restrict__ x, const float* __restrict__ bz,
    const float* __restrict__ br, const float* __restrict__ bh,
    const bf16* __restrict__ wpack, bf16* __restrict__ G) {
  int wave = threadIdx.x >> 5;
  int lane = threadIdx.x & 31;
  size_t wid = (size_t)blockIdx.x * 8 + wave;     // 3 * 2048 * 32 waves
  int ntg = wid & 31;
  int mt  = (wid >> 5) & 2047;
  int g   = (int)(wid >> 16);
  if (g >= 3) return;
  int row0 = mt * 16;
  int n0   = ntg * 16;
  v8f acc = {};
  int r  = lane & 15;
  for (int kt = 0; kt < 16; ++kt) {              // x-part: K = 0..511
    int kk = kt * 32 + ((lane >> 4) << 3);
    const float* p = x + (size_t)(row0 + r) * E_ + kk;
    v16bf a;
#pragma unroll
    for (int i = 0; i < 8; ++i) { a[i] = (bf16)p[i]; a[i + 8] = (bf16)p[i + 16]; }
    const bf16* packp = wpack + ((((size_t)g * NKT + kt) * NNT + ntg) * 32) * 16;
    v16bf b = load_b_pack(packp, lane);
    acc = wmma_bf16(a, b, acc);
  }
  const float* bias = (g == 0) ? bz : (g == 1) ? br : bh;
  bf16* Gg = G + (size_t)g * (size_t)(B_ * T_) * H_;
  int Ml = (lane >> 4) << 3;
  int cc = n0 + (lane & 15);
#pragma unroll
  for (int j = 0; j < 8; ++j) {
    int rr = row0 + j + Ml;                       // flattened (b,t): rr = b*T + t
    int tb = rr & (T_ - 1);                       // t
    int bb = rr >> 9;                             // b  (T_ == 512)
    Gg[((size_t)tb * B_ + bb) * H_ + cc] = (bf16)(acc[j] + bias[cc]);
  }
}

// ---- State init + hidden passthrough ---------------------------------------
__global__ void init_state(const float* __restrict__ h0, float* __restrict__ hf32,
                           bf16* __restrict__ hb) {
  int i = blockIdx.x * 256 + threadIdx.x;
  if (i >= B_ * H_) return;
  float v = h0[i];
  hf32[i] = v; hf32[2 * B_ * H_ + i] = v;         // [dir][pp=0][B*H]
  hb[i] = (bf16)v; hb[2 * B_ * H_ + i] = (bf16)v;
}

__global__ void copy_hidden(const float* __restrict__ h0, float* __restrict__ dst) {
  int i = blockIdx.x * 256 + threadIdx.x;
  if (i < B_ * H_) dst[i] = h0[i];
}

// ---- Device-scope barrier across the 16 WGs of one direction ----------------
__device__ __forceinline__ void dir_barrier(unsigned* ctr, unsigned epoch) {
  __threadfence();
  __syncthreads();
  if (threadIdx.x == 0) {
    __atomic_fetch_add(ctr, 1u, __ATOMIC_RELEASE);
    unsigned target = epoch * NWG_DIR;            // monotonic counter, no reset
    while (__atomic_load_n((volatile unsigned*)ctr, __ATOMIC_ACQUIRE) < target)
      __builtin_amdgcn_s_sleep(1);
  }
  __syncthreads();
  __threadfence();
}

// ---- Sequential scan: both directions, persistent kernel --------------------
// Recurrent weight slice per WG (3 gates x 16 ktiles x 2 ntiles x 1KB = 96KB)
// is staged once into LDS; all inner-loop B-operands come from ds_load_b128.
__global__ void __launch_bounds__(256, 1) gru_scan(
    const bf16* __restrict__ G, const bf16* __restrict__ wpack,
    float* __restrict__ hf32, bf16* __restrict__ hbf16, bf16* __restrict__ rhbuf,
    unsigned* __restrict__ barrier_ctr, float* __restrict__ out) {
  const int dir  = blockIdx.x >> 4;
  const int wg   = blockIdx.x & 15;
  const int wave = threadIdx.x >> 5;
  const int lane = threadIdx.x & 31;
  const int mt   = wave >> 1;                     // 4 M-tiles cover B=64
  const int nt   = wave & 1;                      // 2 N-tiles per WG
  const int ntg  = wg * 2 + nt;
  const int row0 = mt * 16;
  const int col  = ntg * 16 + (lane & 15);
  const int Ml   = (lane >> 4) << 3;

  // ---- one-time LDS staging of this WG's recurrent weights (96 KB) ----
  __shared__ __align__(16) bf16 wlds[3 * 16 * 2 * 512];
  for (int c = threadIdx.x; c < 6144; c += 256) { // 6144 x 16B chunks
    int within = c & 63;
    int tile   = c >> 6;                          // 0..95 = ((g*16+ktl)*2+nt2)
    int nt2    = tile & 1;
    int ktl    = (tile >> 1) & 15;
    int g      = tile >> 5;
    const v4u* src = (const v4u*)(wpack +
        ((((size_t)g * NKT + (16 + ktl)) * NNT + (wg * 2 + nt2)) * 512)) + within;
    ((v4u*)wlds)[(size_t)tile * 64 + within] = *src;
  }
  __syncthreads();

  float* hF = hf32  + (size_t)dir * 2 * B_ * H_;
  bf16*  hB = hbf16 + (size_t)dir * 2 * B_ * H_;
  bf16*  rh = rhbuf + (size_t)dir * B_ * H_;
  unsigned* ctr = barrier_ctr + dir * 32;         // separate cachelines

  const bf16* Gz = G;
  const bf16* Gr = G + (size_t)(B_ * T_) * H_;
  const bf16* Gh = G + 2 * (size_t)(B_ * T_) * H_;

  unsigned epoch = 0;
  int pp = 0;
  for (int tt = 0; tt < T_; ++tt) {
    const int t = dir ? (T_ - 1 - tt) : tt;
    const bf16* hsrc = hB + (size_t)pp * B_ * H_;
    const size_t tbase = (size_t)t * B_ * H_;     // time-major G block for step t

    // Phase 1: z,r gates — h-part GEMM over K=512, weights from LDS
    v8f accz = {}, accr = {};
#pragma unroll 4
    for (int kt = 0; kt < 16; ++kt) {
      v16bf a = load_a_bf16(hsrc, H_, lane, kt * 32);
      const bf16* pz = wlds + (size_t)(((0 * 16 + kt) * 2 + nt) * 512);
      const bf16* pr = wlds + (size_t)(((1 * 16 + kt) * 2 + nt) * 512);
      accz = wmma_bf16(a, load_b_pack(pz, lane), accz);
      accr = wmma_bf16(a, load_b_pack(pr, lane), accr);
    }
    float zv[8], hold[8];
#pragma unroll
    for (int j = 0; j < 8; ++j) {
      int b = row0 + j + Ml;
      size_t gi = tbase + (size_t)b * H_ + col;   // contiguous 64KB block per step
      float z = sigmoidf_(accz[j] + (float)Gz[gi]);
      float r = sigmoidf_(accr[j] + (float)Gr[gi]);
      float h_old = hF[(size_t)pp * B_ * H_ + (size_t)b * H_ + col];
      zv[j] = z; hold[j] = h_old;
      rh[(size_t)b * H_ + col] = (bf16)(r * h_old);
    }
    dir_barrier(ctr, ++epoch);                    // r*h fully published

    // Phase 2: candidate GEMM on r*h, weights from LDS
    v8f acch = {};
#pragma unroll 4
    for (int kt = 0; kt < 16; ++kt) {
      v16bf a = load_a_bf16(rh, H_, lane, kt * 32);
      const bf16* ph = wlds + (size_t)(((2 * 16 + kt) * 2 + nt) * 512);
      acch = wmma_bf16(a, load_b_pack(ph, lane), acch);
    }
    // prefetch next step's precomputed-gate lines while finishing this step
    if (tt + 1 < T_) {
      int t2 = dir ? (T_ - 2 - tt) : (tt + 1);
      size_t g2 = (size_t)t2 * B_ * H_ + (size_t)(row0 + Ml) * H_ + col;
      __builtin_prefetch(Gz + g2, 0, 0);
      __builtin_prefetch(Gr + g2, 0, 0);
      __builtin_prefetch(Gh + g2, 0, 0);
    }
#pragma unroll
    for (int j = 0; j < 8; ++j) {
      int b = row0 + j + Ml;
      size_t gi = tbase + (size_t)b * H_ + col;
      float cand = tanhf(acch[j] + (float)Gh[gi]);
      float hn = (1.0f - zv[j]) * hold[j] + zv[j] * cand;
      hF[(size_t)(pp ^ 1) * B_ * H_ + (size_t)b * H_ + col] = hn;
      hB[(size_t)(pp ^ 1) * B_ * H_ + (size_t)b * H_ + col] = (bf16)hn;
      atomicAdd(&out[((size_t)b * T_ + t) * H_ + col], hn);  // fwd+bwd fused
    }
    dir_barrier(ctr, ++epoch);                    // h_new fully published
    pp ^= 1;
  }
}

// ---- Host-side launch -------------------------------------------------------
extern "C" void kernel_launch(void* const* d_in, const int* in_sizes, int n_in,
                              void* d_out, int out_size, void* d_ws, size_t ws_size,
                              hipStream_t stream) {
  const float* x  = (const float*)d_in[0];
  const float* h0 = (const float*)d_in[1];
  const float* Wz = (const float*)d_in[2];
  const float* bz = (const float*)d_in[3];
  const float* Wr = (const float*)d_in[4];
  const float* br = (const float*)d_in[5];
  const float* Wh = (const float*)d_in[6];
  const float* bh = (const float*)d_in[7];
  float* out = (float*)d_out;

  char* ws = (char*)d_ws;
  unsigned* ctr = (unsigned*)ws;                                  // 256 B
  const size_t wpackN = (size_t)3 << 19;                          // 1,572,864 bf16
  bf16* wpack = (bf16*)(ws + 256);
  const size_t GN = (size_t)3 * B_ * T_ * H_;                     // bf16 x-gates
  bf16* G = (bf16*)(ws + 256 + wpackN * 2);
  char* p = ws + 256 + wpackN * 2 + GN * 2;
  float* hf32 = (float*)p;  p += (size_t)2 * 2 * B_ * H_ * 4;
  bf16*  hb   = (bf16*)p;   p += (size_t)2 * 2 * B_ * H_ * 2;
  bf16*  rh   = (bf16*)p;

  hipMemsetAsync(d_ws, 0, 256, stream);                           // barrier ctrs
  hipMemsetAsync(d_out, 0, (size_t)B_ * T_ * H_ * 4, stream);     // atomics target
  copy_hidden<<<(B_ * H_ + 255) / 256, 256, 0, stream>>>(h0, out + (size_t)B_ * T_ * H_);
  prep_pack<<<(int)((wpackN + 255) / 256), 256, 0, stream>>>(Wz, Wr, Wh, wpack);
  init_state<<<(B_ * H_ + 255) / 256, 256, 0, stream>>>(h0, hf32, hb);
  precompute_x<<<24576, 256, 0, stream>>>(x, bz, br, bh, wpack, G);
  gru_scan<<<32, 256, 0, stream>>>(G, wpack, hf32, hb, rh, ctr, out);
}
